// M_53987738911722
// MI455X (gfx1250) — compile-verified, matrix operand order checked by reading, not run on hardware
//
#include <hip/hip_runtime.h>

// CDNA5 / gfx1250 fused attention tail:
//   probs = softmax(scores + bias) * dropout_keep/0.9 ;  out = probs @ V
// One HBM pass over scores/drop_mask (pass-2 re-read hits L2), V staged in LDS
// as bf16, P@V via v_wmma_f32_16x16x32_bf16 (wave32). Row-pair softmax:
// lanes 0-15 own even row, lanes 16-31 odd row -> 16-lane ds_swizzle reductions.

typedef __attribute__((ext_vector_type(16))) __bf16 v16bf;
typedef __attribute__((ext_vector_type(8)))  float  v8f;

union ABFrag { unsigned int u[8]; v16bf v; };

constexpr int SEQ = 384;   // S (keys & queries)
constexpr int DH  = 64;    // head dim
constexpr int PST = 392;   // padded LDS row stride in halfs (784B, 8B-aligned)
constexpr int NW  = 8;     // waves per block

// ds_swizzle XOR-mode lane exchange (group-of-32, and_mask=0x1f)
template <int XORM>
__device__ __forceinline__ float swz_xor(float v) {
  return __int_as_float(
      __builtin_amdgcn_ds_swizzle(__float_as_int(v), (XORM << 10) | 0x1f));
}

// pack two f32 -> packed bf16 pair (RNE) : 2x add3 + 1x v_perm_b32
__device__ __forceinline__ unsigned int pack_bf16(float f0, float f1) {
  unsigned int u0 = __float_as_uint(f0), u1 = __float_as_uint(f1);
  u0 = u0 + 0x7fffu + ((u0 >> 16) & 1u);
  u1 = u1 + 0x7fffu + ((u1 >> 16) & 1u);
  // result bytes [3,2]=u1 high half, [1,0]=u0 high half
  return __builtin_amdgcn_perm(u1, u0, 0x07060302u);
}
__device__ __forceinline__ unsigned short f2bf(float f) {
  unsigned int u = __float_as_uint(f);
  return (unsigned short)((u + 0x7fffu + ((u >> 16) & 1u)) >> 16);
}

__global__ __launch_bounds__(256) void attn_pv_kernel(
    const float* __restrict__ scores,   // [B,H,S,S]
    const float* __restrict__ vmat,     // [B,H,S,D]
    const float* __restrict__ bias,     // [1,1,1,S]
    const float* __restrict__ dmask,    // [B,H,S,S] uniforms
    float* __restrict__ out)            // [B,H,S,D]
{
  __shared__ unsigned short lds_vT[DH * PST];        // V^T bf16 [n][k], padded
  __shared__ unsigned short lds_p[NW][16 * PST];     // per-wave probs bf16 [m][k]
  __shared__ float lds_bias[SEQ];

  const int tid  = threadIdx.x;
  const int lane = tid & 31;
  const int wave = tid >> 5;
  const int bh   = blockIdx.x;           // 0..B*H-1
  const int qblk = blockIdx.y;           // 0..2
  const size_t head_off = (size_t)bh * SEQ;

  // ---- stage bias ----
  for (int i = tid; i < SEQ; i += 256) lds_bias[i] = bias[i];

  // ---- stage V^T as bf16 into LDS (coalesced global read) ----
  {
    const float* vbase = vmat + head_off * DH;
#pragma unroll
    for (int j = 0; j < (SEQ * DH) / 256; ++j) {
      int i = j * 256 + tid;
      int k = i >> 6, n = i & 63;
      lds_vT[n * PST + k] = f2bf(vbase[i]);
    }
  }
  __syncthreads();

  const int q0    = qblk * 128 + wave * 16;  // this wave's 16 query rows
  const int mrow  = lane & 15;
  const int khalf = lane >> 4;               // 0: even rows, 1: odd rows

  // ---- pass 1: row maxima; lane half owns row 2j+khalf, 16-lane reduce ----
  float rmax8[8];
#pragma unroll
  for (int j = 0; j < 8; ++j) {
    const float4* srow =
        (const float4*)(scores + (head_off + q0 + 2 * j + khalf) * SEQ);
    float v = -3.0e38f;
#pragma unroll
    for (int i = 0; i < 6; ++i) {
      float4 x = srow[i * 16 + mrow];
      const float4 b = *(const float4*)(lds_bias + (i * 16 + mrow) * 4);
      v = fmaxf(v, fmaxf(fmaxf(x.x + b.x, x.y + b.y),
                         fmaxf(x.z + b.z, x.w + b.w)));
    }
    v = fmaxf(v, swz_xor<8>(v));
    v = fmaxf(v, swz_xor<4>(v));
    v = fmaxf(v, swz_xor<2>(v));
    v = fmaxf(v, swz_xor<1>(v));
    rmax8[j] = v;
  }

  // ---- pass 2: exp, row sums, dropout; bf16 probs -> LDS (re-read from L2) ----
  float rcs8[8];   // 1/rowsum for rows 2j+khalf
  unsigned short* pw = lds_p[wave];
#pragma unroll
  for (int j = 0; j < 8; ++j) {
    const size_t roff = (head_off + q0 + 2 * j + khalf) * SEQ;
    const float4* srow = (const float4*)(scores + roff);
    const float4* drow = (const float4*)(dmask + roff);
    float s = 0.f;
#pragma unroll
    for (int i = 0; i < 6; ++i) {
      float4 x = srow[i * 16 + mrow];
      float4 d = drow[i * 16 + mrow];
      int c = (i * 16 + mrow) * 4;
      const float4 b = *(const float4*)(lds_bias + c);
      float e0 = __expf(x.x + b.x - rmax8[j]);
      float e1 = __expf(x.y + b.y - rmax8[j]);
      float e2 = __expf(x.z + b.z - rmax8[j]);
      float e3 = __expf(x.w + b.w - rmax8[j]);
      s += (e0 + e1) + (e2 + e3);              // softmax denom excludes dropout
      float k0 = (d.x > 0.1f) ? (1.0f / 0.9f) : 0.0f;
      float k1 = (d.y > 0.1f) ? (1.0f / 0.9f) : 0.0f;
      float k2 = (d.z > 0.1f) ? (1.0f / 0.9f) : 0.0f;
      float k3 = (d.w > 0.1f) ? (1.0f / 0.9f) : 0.0f;
      uint2 pk;
      pk.x = pack_bf16(e0 * k0, e1 * k1);
      pk.y = pack_bf16(e2 * k2, e3 * k3);
      *(uint2*)(pw + (2 * j + khalf) * PST + c) = pk;   // aligned b64
    }
    s += swz_xor<8>(s);
    s += swz_xor<4>(s);
    s += swz_xor<2>(s);
    s += swz_xor<1>(s);
    rcs8[j] = __builtin_amdgcn_rcpf(s);        // fast v_rcp_f32
  }
  // exchange row sums across lane halves (other parity rows)
  float oth8[8];
#pragma unroll
  for (int j = 0; j < 8; ++j) oth8[j] = swz_xor<16>(rcs8[j]);
  __syncthreads();

  // ---- pass 3: P[16,384] @ V[384,64] via v_wmma_f32_16x16x32_bf16 ----
  v8f acc[4] = {v8f{}, v8f{}, v8f{}, v8f{}};
#pragma unroll
  for (int kt = 0; kt < 12; ++kt) {
    ABFrag a;
#pragma unroll
    for (int v = 0; v < 8; ++v) {
      // CDNA5 16-bit A/B layout: K = 16*(v/4) + 8*lanehalf + 2*(v%4) (+pair)
      int k = kt * 32 + ((v >> 2) << 4) + (khalf << 3) + ((v & 3) << 1);
      a.u[v] = *(const unsigned int*)(pw + mrow * PST + k);
    }
    ABFrag b4[4];
#pragma unroll
    for (int nt = 0; nt < 4; ++nt) {
#pragma unroll
      for (int v = 0; v < 8; ++v) {
        int k = kt * 32 + ((v >> 2) << 4) + (khalf << 3) + ((v & 3) << 1);
        b4[nt].u[v] =
            *(const unsigned int*)(lds_vT + (nt * 16 + mrow) * PST + k);
      }
    }
#pragma unroll
    for (int nt = 0; nt < 4; ++nt) {
      acc[nt] = __builtin_amdgcn_wmma_f32_16x16x32_bf16(
          false, a.v, false, b4[nt].v, (short)0, acc[nt], false, false);
    }
  }

  // ---- epilogue: C/D layout lane=N, VGPR r -> M = r + 8*lanehalf ----
  // Row m = r + 8*khalf has parity (r&1); lane owns parity khalf ->
  // constant-indexed selects only, no VGPR spills.
  float myrcs[8];
#pragma unroll
  for (int r = 0; r < 8; ++r) {
    const int jj = r >> 1;
    float a0 = rcs8[jj], a1 = rcs8[jj + 4];
    float b0 = oth8[jj], b1 = oth8[jj + 4];
    myrcs[r] = (r & 1) ? (khalf ? a1 : b0) : (khalf ? b1 : a0);
  }

  float* obase = out + (head_off + q0) * DH;
#pragma unroll
  for (int nt = 0; nt < 4; ++nt) {
#pragma unroll
    for (int r = 0; r < 8; ++r) {
      int m = r + 8 * khalf;
      obase[(size_t)m * DH + nt * 16 + mrow] = acc[nt][r] * myrcs[r];
    }
  }
}

extern "C" void kernel_launch(void* const* d_in, const int* in_sizes, int n_in,
                              void* d_out, int out_size, void* d_ws, size_t ws_size,
                              hipStream_t stream) {
  const float* x91   = (const float*)d_in[0];   // [32,12,384,384] scores
  const float* x83   = (const float*)d_in[1];   // [32,12,384,64]  values
  const float* bias  = (const float*)d_in[2];   // [384]
  const float* dmask = (const float*)d_in[3];   // [32,12,384,384] uniforms
  float* out = (float*)d_out;                   // [32,12,384,64]

  dim3 grid(32 * 12, 3, 1);   // (head-batch, query block of 128)
  dim3 block(256, 1, 1);      // 8 wave32 per block
  attn_pv_kernel<<<grid, block, 0, stream>>>(x91, x83, bias, dmask, out);
}
